// ANPCrossAttentionLayer_36009005810095
// MI455X (gfx1250) — matrix operand order repaired and checked by reading, not compiled, not run on hardware
//
#include <hip/hip_runtime.h>
#include <hip/hip_bf16.h>

typedef __attribute__((ext_vector_type(16))) _Float16 v16h;
typedef __attribute__((ext_vector_type(8)))  _Float16 v8h;
typedef __attribute__((ext_vector_type(8)))  float    v8f;

#define HH 16
#define LL 8192
#define SS 512
#define DD 1024
#define BB 2
#define DH 64

// ---- WMMA helpers -----------------------------------------------------------

__device__ __forceinline__ v8f wmma_f16(v16h a, v16h b, v8f c) {
  // v_wmma_f32_16x16x32_f16  D = A(16x32) x B(32x16) + C
  return __builtin_amdgcn_wmma_f32_16x16x32_f16(false, a, false, b, (short)0, c,
                                                false, false);
}

// A fragment: lane holds row `row`, halves[0..7] = K k0+hi*8.., halves[8..15] = K k0+16+hi*8..
__device__ __forceinline__ v16h load_frag_a(const _Float16* __restrict__ base,
                                            int ldk, int row, int k0, int hi) {
  const _Float16* p = base + (size_t)row * ldk + k0 + hi * 8;
  v16h r;
  ((v8h*)&r)[0] = *(const v8h*)(p);
  ((v8h*)&r)[1] = *(const v8h*)(p + 16);
  return r;
}

// B fragment (col-major column == row of W / row of Vt): lane holds column `col`,
// halves[0..7] = K k0+hi*8.., halves[8..15] = K k0+16+hi*8..
__device__ __forceinline__ v16h load_frag_b(const _Float16* __restrict__ base,
                                            int ldk, int col, int k0, int hi) {
  const _Float16* p = base + (size_t)col * ldk + k0 + hi * 8;
  v16h r;
  ((v8h*)&r)[0] = *(const v8h*)(p);
  ((v8h*)&r)[1] = *(const v8h*)(p + 16);
  return r;
}

// ---- small utility kernels --------------------------------------------------

__global__ void zero_kernel(float* p, int n) {
  int i = blockIdx.x * blockDim.x + threadIdx.x;
  if (i < n) p[i] = 0.0f;
}

__global__ void f2h_kernel(const float* __restrict__ x, _Float16* __restrict__ y, int n) {
  int i = blockIdx.x * blockDim.x + threadIdx.x;
  int stride = gridDim.x * blockDim.x;
  for (; i < n; i += stride) y[i] = (_Float16)x[i];
}

// LayerNorm: one 256-thread block per row of D=1024, f32 in -> f16 out.
__global__ __launch_bounds__(256) void ln_kernel(const float* __restrict__ x,
                                                 const float* __restrict__ w,
                                                 const float* __restrict__ bsh,
                                                 _Float16* __restrict__ y) {
  __shared__ float s1[256], s2[256];
  const int row = blockIdx.x;
  const float* xr = x + (size_t)row * DD;
  float sum = 0.f, sq = 0.f;
  for (int i = threadIdx.x; i < DD; i += 256) {
    float v = xr[i];
    sum += v; sq += v * v;
  }
  s1[threadIdx.x] = sum; s2[threadIdx.x] = sq;
  __syncthreads();
  for (int o = 128; o > 0; o >>= 1) {
    if (threadIdx.x < o) {
      s1[threadIdx.x] += s1[threadIdx.x + o];
      s2[threadIdx.x] += s2[threadIdx.x + o];
    }
    __syncthreads();
  }
  const float mu = s1[0] * (1.0f / DD);
  const float var = s2[0] * (1.0f / DD) - mu * mu;
  const float rstd = rsqrtf(var + 1e-6f);
  _Float16* yr = y + (size_t)row * DD;
  for (int i = threadIdx.x; i < DD; i += 256)
    yr[i] = (_Float16)((xr[i] - mu) * rstd * w[i] + bsh[i]);
}

// ---- WMMA GEMM: C[M,N] = A[M,K=1024] @ W[N,K=1024]^T -----------------------
// Each wave owns a 16x64 strip of C (4 accumulators). Modulo-4 buffer ring,
// prefetch distance 2, prefetch issued BEFORE the compute of each stage so
// the per-stage wait is a partial s_wait_loadcnt (two bursts outstanding).
// #pragma unroll 4 constant-folds the ring indices (trip count 32 % 4 == 0).
// mode 0: f16 row-major out   (Q, K, attn-out)
// mode 1: f16 transposed out  (V -> Vt[b,h,d,s]); sdim = S
// mode 2: f32 out, (c + bias[n]) * gate
__global__ __launch_bounds__(256) void gemm16_wmma(const _Float16* __restrict__ A,
                                                   const _Float16* __restrict__ W,
                                                   int M, int N,
                                                   int mode, int sdim,
                                                   _Float16* __restrict__ outh,
                                                   float* __restrict__ outf,
                                                   const float* __restrict__ bias,
                                                   const float* __restrict__ gate) {
  const int wave = (blockIdx.x * blockDim.x + threadIdx.x) >> 5;
  const int lane = threadIdx.x & 31;
  const int lo = lane & 15, hi = lane >> 4;
  const int n64 = N >> 6;
  const int mt = wave / n64;
  const int nt = wave - mt * n64;
  if (mt * 16 >= M) return;
  const int m0 = mt * 16;
  const int nb = nt * 64;

  constexpr int NK = DD / 32;   // 32 k-chunks

  v8f acc[4] = {v8f{}, v8f{}, v8f{}, v8f{}};

  v16h a[4];
  v16h bf[4][4];
#pragma unroll
  for (int s = 0; s < 2; ++s) {
    a[s] = load_frag_a(A, DD, m0 + lo, s * 32, hi);
#pragma unroll
    for (int j = 0; j < 4; ++j)
      bf[s][j] = load_frag_b(W, DD, nb + j * 16 + lo, s * 32, hi);
  }

#pragma unroll 4
  for (int kc = 0; kc < NK; ++kc) {
    const int cur = kc & 3;
    const int pre = (kc + 2) & 3;
    if (kc + 2 < NK) {
      const int kn = (kc + 2) * 32;
      a[pre] = load_frag_a(A, DD, m0 + lo, kn, hi);
#pragma unroll
      for (int j = 0; j < 4; ++j)
        bf[pre][j] = load_frag_b(W, DD, nb + j * 16 + lo, kn, hi);
    }
#pragma unroll
    for (int j = 0; j < 4; ++j) acc[j] = wmma_f16(a[cur], bf[cur][j], acc[j]);
  }

  const float g = (mode == 2) ? gate[0] : 1.0f;
#pragma unroll
  for (int j = 0; j < 4; ++j) {
    const int n = nb + j * 16 + lo;
#pragma unroll
    for (int r = 0; r < 8; ++r) {
      const int m = m0 + r + hi * 8;
      const float v = acc[j][r];
      if (mode == 0) {
        outh[(size_t)m * N + n] = (_Float16)v;
      } else if (mode == 1) {
        // Vt[b, n, s]: b = m / sdim, s = m % sdim
        const size_t idx = (size_t)(m / sdim) * ((size_t)sdim * N) +
                           (size_t)n * sdim + (size_t)(m % sdim);
        outh[idx] = (_Float16)v;
      } else {
        outf[(size_t)m * N + n] = (v + bias[n]) * g;
      }
    }
  }
}

// ---- Attention: one workgroup (4 waves) per (b, h, 16-row L tile) ----------
#define LGS 520   // logits LDS stride (floats)
#define PS  520   // P LDS stride (halves)

__global__ __launch_bounds__(128) void attn_wmma(const _Float16* __restrict__ Q,
                                                 const _Float16* __restrict__ Kh,
                                                 const _Float16* __restrict__ Vt,
                                                 _Float16* __restrict__ Oa,
                                                 float* __restrict__ ent_acc,
                                                 float* __restrict__ top_acc) {
  __shared__ float   Lg[16 * LGS];
  __shared__ _Float16 P[16 * PS];
  __shared__ float   redmax[16 * 8], redZ[16 * 8], redSt[16 * 8];
  __shared__ float   redT4[16 * 8 * 4];
  __shared__ float   rowmax[16], rowZ[16], rowEnt[16], rowTop[16];

  const int tid  = threadIdx.x;
  const int wave = tid >> 5;
  const int lane = tid & 31;
  const int lo = lane & 15, hi = lane >> 4;

  const int b  = blockIdx.x / (HH * (LL / 16));
  const int h  = (blockIdx.x / (LL / 16)) % HH;
  const int lt = blockIdx.x % (LL / 16);
  const int m0    = b * LL + lt * 16;   // row into Q[B*L, D]
  const int krow0 = b * SS;             // row base into K[B*S, D]
  const int doff  = h * DH;
  const float scale = 0.125f;           // 64^-0.5

  // Phase 1: logits[16 x 512] via WMMA (K = d = 64 -> 2 chunks). Modulo-4
  // ring on the S tiles, prefetch distance 2, loads ahead of compute.
  v16h aq0 = load_frag_a(Q + doff, DD, m0 + lo, 0, hi);
  v16h aq1 = load_frag_a(Q + doff, DD, m0 + lo, 32, hi);
  {
    const int sbase = wave * 128;
    v16h bk[4][2];
#pragma unroll
    for (int s = 0; s < 2; ++s) {
      bk[s][0] = load_frag_b(Kh + doff, DD, krow0 + sbase + s * 16 + lo, 0, hi);
      bk[s][1] = load_frag_b(Kh + doff, DD, krow0 + sbase + s * 16 + lo, 32, hi);
    }
#pragma unroll 4
    for (int st = 0; st < 8; ++st) {
      const int cur = st & 3;
      const int pre = (st + 2) & 3;
      const int s0 = sbase + st * 16;
      if (st + 2 < 8) {
        bk[pre][0] = load_frag_b(Kh + doff, DD, krow0 + s0 + 32 + lo, 0, hi);
        bk[pre][1] = load_frag_b(Kh + doff, DD, krow0 + s0 + 32 + lo, 32, hi);
      }
      v8f acc = {};
      acc = wmma_f16(aq0, bk[cur][0], acc);
      acc = wmma_f16(aq1, bk[cur][1], acc);
#pragma unroll
      for (int r = 0; r < 8; ++r)
        Lg[(r + hi * 8) * LGS + s0 + lo] = acc[r] * scale;
    }
  }
  __syncthreads();

  // Phase 2: softmax stats (+ entropy, top-4). 8 threads per row, 64 cols each.
  const int row = tid >> 3, sub = tid & 7;
  {
    const float* lr = &Lg[row * LGS + sub * 64];
    float mx = -3.4e38f;
    for (int i = 0; i < 64; ++i) mx = fmaxf(mx, lr[i]);
    redmax[row * 8 + sub] = mx;
  }
  __syncthreads();
  if (sub == 0) {
    float mx = redmax[row * 8];
    for (int i = 1; i < 8; ++i) mx = fmaxf(mx, redmax[row * 8 + i]);
    rowmax[row] = mx;
  }
  __syncthreads();
  {
    const float mx = rowmax[row];
    const float* lr = &Lg[row * LGS + sub * 64];
    float z = 0.f, st = 0.f;
    float t4[4] = {-3.4e38f, -3.4e38f, -3.4e38f, -3.4e38f};
    for (int i = 0; i < 64; ++i) {
      const float t = lr[i] - mx;
      const float e = __expf(t);
      z += e; st += t * e;
      if (t > t4[3]) {
        if      (t > t4[0]) { t4[3]=t4[2]; t4[2]=t4[1]; t4[1]=t4[0]; t4[0]=t; }
        else if (t > t4[1]) { t4[3]=t4[2]; t4[2]=t4[1]; t4[1]=t; }
        else if (t > t4[2]) { t4[3]=t4[2]; t4[2]=t; }
        else                { t4[3]=t; }
      }
    }
    redZ[row * 8 + sub] = z; redSt[row * 8 + sub] = st;
#pragma unroll
    for (int i = 0; i < 4; ++i) redT4[(row * 8 + sub) * 4 + i] = t4[i];
  }
  __syncthreads();
  if (sub == 0) {
    float z = 0.f, st = 0.f;
    float t4[4] = {-3.4e38f, -3.4e38f, -3.4e38f, -3.4e38f};
    for (int i = 0; i < 8; ++i) {
      z  += redZ[row * 8 + i];
      st += redSt[row * 8 + i];
      for (int q2 = 0; q2 < 4; ++q2) {
        const float t = redT4[(row * 8 + i) * 4 + q2];
        if (t > t4[3]) {
          if      (t > t4[0]) { t4[3]=t4[2]; t4[2]=t4[1]; t4[1]=t4[0]; t4[0]=t; }
          else if (t > t4[1]) { t4[3]=t4[2]; t4[2]=t4[1]; t4[1]=t; }
          else if (t > t4[2]) { t4[3]=t4[2]; t4[2]=t; }
          else                { t4[3]=t; }
        }
      }
    }
    rowZ[row]   = z;
    rowEnt[row] = __logf(z) - st / z;   // -(sum p log p)
    rowTop[row] = (__expf(t4[0]) + __expf(t4[1]) + __expf(t4[2]) + __expf(t4[3])) / z;
  }
  __syncthreads();
  if (tid == 0) {
    float es = 0.f, ts = 0.f;
    for (int r = 0; r < 16; ++r) { es += rowEnt[r]; ts += rowTop[r]; }
    atomicAdd(&ent_acc[h], es);
    atomicAdd(top_acc, ts);
  }

  // Phase 3: P = softmax(logits) as f16 into LDS
  {
    const float mx = rowmax[row];
    const float invz = 1.0f / rowZ[row];
    const float* lr = &Lg[row * LGS + sub * 64];
    _Float16*    pr = &P[row * PS + sub * 64];
    for (int i = 0; i < 64; ++i)
      pr[i] = (_Float16)(__expf(lr[i] - mx) * invz);
  }
  __syncthreads();

  // Phase 4: out[16 x 64] = P[16 x 512] @ V[512 x 64]; wave owns one 16-wide
  // d tile. Modulo-4 ring on the Vt fragments; A from LDS (ds_load_b128).
  {
    const int n0 = wave * 16;
    const int vrow0 = (b * HH + h) * DH + n0;     // rows of Vt[b,h,d, s]
    const _Float16* Pp = &P[0];
    v8f acc = {};
    v16h bv[4];
    bv[0] = load_frag_b(Vt, SS, vrow0 + lo, 0, hi);
    bv[1] = load_frag_b(Vt, SS, vrow0 + lo, 32, hi);
#pragma unroll 4
    for (int kc = 0; kc < 16; ++kc) {
      const int cur = kc & 3;
      const int pre = (kc + 2) & 3;
      if (kc + 2 < 16)
        bv[pre] = load_frag_b(Vt, SS, vrow0 + lo, (kc + 2) * 32, hi);
      v16h a = load_frag_a(Pp, PS, lo, kc * 32, hi);
      acc = wmma_f16(a, bv[cur], acc);
    }
#pragma unroll
    for (int r = 0; r < 8; ++r) {
      const int m = m0 + r + hi * 8;
      Oa[(size_t)m * DD + doff + n0 + lo] = (_Float16)acc[r];
    }
  }
}

__global__ void finalize_diag(const float* __restrict__ diag,
                              const float* __restrict__ gate,
                              float* __restrict__ outp) {
  if (threadIdx.x == 0 && blockIdx.x == 0) {
    float eh[HH];
    float mean = 0.f;
    for (int h = 0; h < HH; ++h) {
      eh[h] = diag[h] / (float)(BB * LL);
      mean += eh[h];
    }
    mean /= (float)HH;
    float var = 0.f;
    for (int h = 0; h < HH; ++h) { const float d = eh[h] - mean; var += d * d; }
    var /= (float)HH;
    outp[0] = mean;
    outp[1] = sqrtf(var);
    outp[2] = diag[HH] / (float)(BB * HH * LL);
    outp[3] = gate[0];
  }
}

// ---- launch ----------------------------------------------------------------

extern "C" void kernel_launch(void* const* d_in, const int* in_sizes, int n_in,
                              void* d_out, int out_size, void* d_ws, size_t ws_size,
                              hipStream_t stream) {
  const float* q_in  = (const float*)d_in[0];
  const float* kv_in = (const float*)d_in[1];
  const float* qn_w  = (const float*)d_in[2];
  const float* qn_b  = (const float*)d_in[3];
  const float* kvn_w = (const float*)d_in[4];
  const float* kvn_b = (const float*)d_in[5];
  const float* Wq    = (const float*)d_in[6];
  const float* Wk    = (const float*)d_in[7];
  const float* Wv    = (const float*)d_in[8];
  const float* Wo    = (const float*)d_in[9];
  const float* bo    = (const float*)d_in[10];
  const float* gate  = (const float*)d_in[11];

  char* ws = (char*)d_ws;
  const size_t SZ_QX  = (size_t)BB * LL * DD * 2;   // 32 MB
  const size_t SZ_KVX = (size_t)BB * SS * DD * 2;   // 2 MB
  const size_t SZ_W   = (size_t)DD * DD * 2;        // 2 MB

  _Float16* qx  = (_Float16*)(ws);
  _Float16* kvx = (_Float16*)(ws + SZ_QX);
  _Float16* Wqh = (_Float16*)(ws + SZ_QX + SZ_KVX);
  _Float16* Wkh = (_Float16*)(ws + SZ_QX + SZ_KVX + SZ_W);
  _Float16* Wvh = (_Float16*)(ws + SZ_QX + SZ_KVX + 2 * SZ_W);
  _Float16* Woh = (_Float16*)(ws + SZ_QX + SZ_KVX + 3 * SZ_W);
  _Float16* Qh  = (_Float16*)(ws + SZ_QX + SZ_KVX + 4 * SZ_W);
  _Float16* Kh  = (_Float16*)(ws + 2 * SZ_QX + SZ_KVX + 4 * SZ_W);
  _Float16* Vth = (_Float16*)(ws + 2 * SZ_QX + 2 * SZ_KVX + 4 * SZ_W);
  float*    diag = (float*)  (ws + 2 * SZ_QX + 3 * SZ_KVX + 4 * SZ_W);
  _Float16* Oa  = qx;   // alias: qx is dead after the Q projection

  const int M_Q = BB * LL;   // 16384
  const int M_K = BB * SS;   // 1024

  zero_kernel<<<1, 32, 0, stream>>>(diag, HH + 1);
  ln_kernel<<<M_Q, 256, 0, stream>>>(q_in, qn_w, qn_b, qx);
  ln_kernel<<<M_K, 256, 0, stream>>>(kv_in, kvn_w, kvn_b, kvx);
  f2h_kernel<<<512, 256, 0, stream>>>(Wq, Wqh, DD * DD);
  f2h_kernel<<<512, 256, 0, stream>>>(Wk, Wkh, DD * DD);
  f2h_kernel<<<512, 256, 0, stream>>>(Wv, Wvh, DD * DD);
  f2h_kernel<<<512, 256, 0, stream>>>(Wo, Woh, DD * DD);

  // Q = qx @ Wq^T  -> f16 [16384, 1024]
  {
    int waves = (M_Q / 16) * (DD / 64);
    gemm16_wmma<<<waves / 8, 256, 0, stream>>>(qx, Wqh, M_Q, DD, 0, 0,
                                               Qh, nullptr, nullptr, nullptr);
  }
  // K = kvx @ Wk^T -> f16 [1024, 1024]
  {
    int waves = (M_K / 16) * (DD / 64);
    gemm16_wmma<<<waves / 8, 256, 0, stream>>>(kvx, Wkh, M_K, DD, 0, 0,
                                               Kh, nullptr, nullptr, nullptr);
  }
  // Vt = transpose(kvx @ Wv^T) -> f16 [B*H*64, 512]
  {
    int waves = (M_K / 16) * (DD / 64);
    gemm16_wmma<<<waves / 8, 256, 0, stream>>>(kvx, Wvh, M_K, DD, 1, SS,
                                               Vth, nullptr, nullptr, nullptr);
  }
  // attention: 16384 workgroups x 128 threads
  attn_wmma<<<BB * HH * (LL / 16), 128, 0, stream>>>(Qh, Kh, Vth, Oa,
                                                     diag, diag + HH);
  // out = (Oa @ Wo^T + bo) * gate -> f32 d_out
  {
    int waves = (M_Q / 16) * (DD / 64);
    gemm16_wmma<<<waves / 8, 256, 0, stream>>>(Oa, Woh, M_Q, DD, 2, 0,
                                               nullptr, (float*)d_out, bo, gate);
  }
  finalize_diag<<<1, 1, 0, stream>>>(diag, gate, (float*)d_out + (size_t)BB * LL * DD);
}
